// WindowAttention_44478681318218
// MI455X (gfx1250) — matrix operand order repaired and checked by reading, not compiled, not run on hardware
//
#include <hip/hip_runtime.h>
#include <hip/hip_fp16.h>

typedef _Float16     v16h __attribute__((ext_vector_type(16)));
typedef float        v8f  __attribute__((ext_vector_type(8)));
typedef unsigned int v4u  __attribute__((ext_vector_type(4)));
typedef int          v4i  __attribute__((ext_vector_type(4)));

#if __has_builtin(__builtin_amdgcn_global_load_async_to_lds_b128) && \
    __has_builtin(__builtin_amdgcn_s_wait_asynccnt)
#define SWIN_HAS_ASYNC 1
#endif

namespace {
constexpr int BATCH = 2, IMH = 384, IMW = 384;
constexpr int WSZ = 8, DSP = 4, HDIM = 8;
constexpr int NWH = IMH / WSZ, NWW = IMW / WSZ;   // 48 x 48 windows

union F16Frag { v16h h; v4u u[2]; _Float16 e[16]; };

__device__ inline v4u vz4() { v4u z = {0u, 0u, 0u, 0u}; return z; }

// Cooperative 16B-granule copy global -> LDS (async path when available).
__device__ inline void stage_to_lds(const void* src, void* dst, int n16,
                                    int tid, int nthreads)
{
#ifdef SWIN_HAS_ASYNC
  v4i* s = (v4i*)src;   // C-style cast drops const
  v4i* d = (v4i*)dst;
  for (int i = tid; i < n16; i += nthreads)
    __builtin_amdgcn_global_load_async_to_lds_b128(
        (__attribute__((address_space(1))) v4i*)(s + i),
        (__attribute__((address_space(3))) v4i*)(d + i), 0, 0);
  __builtin_amdgcn_s_wait_asynccnt(0);
#else
  const v4u* s = (const v4u*)src;
  v4u*       d = (v4u*)dst;
  for (int i = tid; i < n16; i += nthreads) d[i] = s[i];
#endif
}
}

// ---------------------------------------------------------------------------
// Zero a small pad region used for branchless SAME-padding loads.
// ---------------------------------------------------------------------------
__global__ void swin_zero_kernel(float* __restrict__ p)
{
  p[threadIdx.x] = 0.f;
}

// ---------------------------------------------------------------------------
// conv1: 3 -> 64, 3x3 SAME, ReLU, fp32 math, f16 NHWC output (K=27: VALU).
// Weights staged in LDS once per block. All in[] loops unrolled so the
// 27-entry stencil stays in VGPRs.
// ---------------------------------------------------------------------------
__global__ void swin_conv1_kernel(const float* __restrict__ x,   // (B,3,H,W)
                                  const float* __restrict__ w,   // (64,3,3,3)
                                  const float* __restrict__ bia, // (64)
                                  _Float16* __restrict__ out)    // (B,H,W,64)
{
  __shared__ float ws[64 * 27];
  __shared__ float bs[64];
  for (int i = threadIdx.x; i < 64 * 27; i += blockDim.x) ws[i] = w[i];
  if (threadIdx.x < 64) bs[threadIdx.x] = bia[threadIdx.x];
  __syncthreads();

  int idx = blockIdx.x * blockDim.x + threadIdx.x;
  if (idx >= BATCH * IMH * IMW) return;
  int xw = idx % IMW, yh = (idx / IMW) % IMH, b = idx / (IMW * IMH);

  float in[3][3][3];
#pragma unroll
  for (int ci = 0; ci < 3; ++ci)
#pragma unroll
    for (int ky = 0; ky < 3; ++ky) {
      int yy = yh + ky - 1;
#pragma unroll
      for (int kx = 0; kx < 3; ++kx) {
        int xx = xw + kx - 1;
        float v = 0.f;
        if (yy >= 0 && yy < IMH && xx >= 0 && xx < IMW)
          v = x[((size_t)(b * 3 + ci) * IMH + yy) * IMW + xx];
        in[ci][ky][kx] = v;
      }
    }
  _Float16* op = out + ((size_t)(b * IMH + yh) * IMW + xw) * 64;
  for (int co = 0; co < 64; ++co) {
    float acc = bs[co];
    const float* wp = &ws[co * 27];
#pragma unroll
    for (int ci = 0; ci < 3; ++ci)
#pragma unroll
      for (int ky = 0; ky < 3; ++ky)
#pragma unroll
        for (int kx = 0; kx < 3; ++kx)
          acc = fmaf(in[ci][ky][kx], wp[(ci * 3 + ky) * 3 + kx], acc);
    op[co] = (_Float16)fmaxf(acc, 0.f);
  }
}

// ---------------------------------------------------------------------------
// Weight repack: OIHW fp32 (32, cin, 3, 3) -> WMMA B fragments
// (lanes 0-15: K=0-15, col=lane; lanes 16-31: K=16-31, col=lane-16),
// 16 f16 contiguous per lane, tiles ordered [tap][kc][nt].
// ---------------------------------------------------------------------------
__global__ void swin_repack_kernel(const float* __restrict__ w, int cin,
                                   _Float16* __restrict__ out)
{
  int lane = threadIdx.x & 31;
  int tile = blockIdx.x;
  int kcn = cin >> 5;
  int nt  = tile & 1;
  int kc  = (tile >> 1) % kcn;
  int tap = tile / (2 * kcn);
  int ky = tap / 3, kx = tap % 3;
  int n = lane & 15;
  int kbase = (lane < 16) ? 0 : 16;
  int o = nt * 16 + n;
  _Float16* op = out + (size_t)tile * 512 + lane * 16;
#pragma unroll
  for (int i = 0; i < 16; ++i) {
    int k = kc * 32 + kbase + i;
    op[i] = (_Float16)w[(((size_t)o * cin + k) * 3 + ky) * 3 + kx];
  }
}

// ---------------------------------------------------------------------------
// Implicit-GEMM 3x3 conv with v_wmma_f32_16x16x32_f16, templated on CIN so
// the 9-tap x kchunk loop fully unrolls. Weights staged once per block in
// LDS (async copy); A loads are branchless via zero-pad pointer select.
// One wave computes a 16-pixel x 32-channel output tile.
// ---------------------------------------------------------------------------
template<int CIN, int RELU>
__global__ void swin_convw_kernel(const _Float16* __restrict__ in, // (B,H,W,CIN)
                                  const _Float16* __restrict__ wp, // packed B tiles
                                  const float* __restrict__ bia,   // (32)
                                  _Float16* __restrict__ out,      // (B,H,W,32)
                                  const _Float16* __restrict__ zpad)
{
  constexpr int KCN = CIN / 32;
  constexpr int NTILES = 9 * KCN * 2;
  __shared__ _Float16 Wsm[NTILES * 512];

  stage_to_lds(wp, Wsm, NTILES * 64, threadIdx.x, blockDim.x);
  __syncthreads();

  int wave = threadIdx.x >> 5;
  int lane = threadIdx.x & 31;
  int xt = blockIdx.x * (blockDim.x >> 5) + wave;   // 0..23
  int y = blockIdx.y, b = blockIdx.z;
  int x0 = xt * 16;
  int half = lane >> 4, ln = lane & 15;

  v8f acc0 = {0.f,0.f,0.f,0.f,0.f,0.f,0.f,0.f};
  v8f acc1 = {0.f,0.f,0.f,0.f,0.f,0.f,0.f,0.f};

#pragma unroll
  for (int tap = 0; tap < 9; ++tap) {
    const int ky = tap / 3, kx = tap % 3;
    int yy = y + ky - 1;
    int xx = x0 + ln + kx - 1;
    bool ok = (yy >= 0) & (yy < IMH) & (xx >= 0) & (xx < IMW);
    const _Float16* rowp = in + ((size_t)(b * IMH + yy) * IMW + xx) * CIN;
#pragma unroll
    for (int kc = 0; kc < KCN; ++kc) {
      const _Float16* ip = ok ? (rowp + kc * 32 + half * 16) : zpad;
      F16Frag A;
      A.u[0] = *(const v4u*)(ip);
      A.u[1] = *(const v4u*)(ip + 8);
      const _Float16* wt = &Wsm[((tap * KCN + kc) * 2) * 512 + lane * 16];
      F16Frag B0, B1;
      B0.u[0] = *(const v4u*)(wt);
      B0.u[1] = *(const v4u*)(wt + 8);
      B1.u[0] = *(const v4u*)(wt + 512);
      B1.u[1] = *(const v4u*)(wt + 520);
      acc0 = __builtin_amdgcn_wmma_f32_16x16x32_f16(false, A.h, false, B0.h,
                                                    (short)0, acc0, false, false);
      acc1 = __builtin_amdgcn_wmma_f32_16x16x32_f16(false, A.h, false, B1.h,
                                                    (short)0, acc1, false, false);
    }
  }
  // C layout: row = r + 8*half (pixel), col = ln (channel within ntile)
  float bi0 = bia[ln], bi1 = bia[16 + ln];
  _Float16* op = out + ((size_t)(b * IMH + y) * IMW + x0) * 32;
#pragma unroll
  for (int r = 0; r < 8; ++r) {
    int px = r + 8 * half;
    float v0 = acc0[r] + bi0;
    float v1 = acc1[r] + bi1;
    if (RELU) { v0 = fmaxf(v0, 0.f); v1 = fmaxf(v1, 0.f); }
    op[(size_t)px * 32 + ln]      = (_Float16)v0;
    op[(size_t)px * 32 + 16 + ln] = (_Float16)v1;
  }
}

// ---------------------------------------------------------------------------
// Shifted-window attention. Block = 128 threads = 4 waves = 4 heads of one
// window. Strip-mined over 16-row strips so every register array has
// constant indices (no scratch spills): per strip, 4 QK^T WMMAs -> softmax
// -> exp(P) to LDS; then per strip 2 P*V WMMAs -> store. 1/rowsum folded
// into the store. Output written in the SHIFTED frame.
// ---------------------------------------------------------------------------
__global__ void swin_attn_kernel(const _Float16* __restrict__ qb,
                                 const _Float16* __restrict__ kb,
                                 const _Float16* __restrict__ vb,
                                 const float* __restrict__ pos_emb, // (15,15)
                                 _Float16* __restrict__ outb,       // (B,H,W,32)
                                 const _Float16* __restrict__ zpad)
{
  __shared__ _Float16 Vs[64 * 32];
  __shared__ _Float16 Ps[4][64 * 72];
  __shared__ float pe[225];

  for (int i = threadIdx.x; i < 225; i += blockDim.x) pe[i] = pos_emb[i];

  int wave = threadIdx.x >> 5;       // head
  int lane = threadIdx.x & 31;
  int win = blockIdx.x, b = blockIdx.y;
  int wh = win / NWW, ww = win % NWW;
  int half = lane >> 4, ln = lane & 15;
  int hc = wave * HDIM;

  // Stage V (64 tokens x 32 ch) cooperatively: each thread 16 f16.
  {
    int t = threadIdx.x >> 1;
    int c = (threadIdx.x & 1) * 16;
    int wy = t >> 3, wx = t & 7;
    int ys = (wh * WSZ + wy + DSP) % IMH;
    int xs = (ww * WSZ + wx + DSP) % IMW;
    const _Float16* vp = vb + ((size_t)(b * IMH + ys) * IMW + xs) * 32 + c;
    *(v4u*)(&Vs[t * 32 + c])     = *(const v4u*)(vp);
    *(v4u*)(&Vs[t * 32 + c + 8]) = *(const v4u*)(vp + 8);
  }
  __syncthreads();

  const float SC = 0.35355339059327373f;   // 8^-0.5
  bool maskRow = (wh == NWH - 1);
  bool maskCol = (ww == NWW - 1);
  _Float16* P = &Ps[wave][0];

  // K B-fragments for all four 16-column tiles (branchless pad loads).
  F16Frag Bk[4];
#pragma unroll
  for (int ni = 0; ni < 4; ++ni) {
    int t = ni * 16 + ln;
    int wy = t >> 3, wx = t & 7;
    int ys = (wh * WSZ + wy + DSP) % IMH;
    int xs = (ww * WSZ + wx + DSP) % IMW;
    const _Float16* kp = (half == 0)
        ? (kb + ((size_t)(b * IMH + ys) * IMW + xs) * 32 + hc) : zpad;
    Bk[ni].u[0] = *(const v4u*)(kp);
    Bk[ni].u[1] = vz4();
  }

  float rinv[4][8];
#pragma unroll
  for (int mi = 0; mi < 4; ++mi) {
    // Q A-fragment for this 16-row strip.
    int t = mi * 16 + ln;
    int wy = t >> 3, wx = t & 7;
    int ys = (wh * WSZ + wy + DSP) % IMH;
    int xs = (ww * WSZ + wx + DSP) % IMW;
    const _Float16* qp = (half == 0)
        ? (qb + ((size_t)(b * IMH + ys) * IMW + xs) * 32 + hc) : zpad;
    F16Frag Aq;
    Aq.u[0] = *(const v4u*)(qp);
    Aq.u[1] = vz4();

    v8f acc[4];
#pragma unroll
    for (int ni = 0; ni < 4; ++ni) {
      v8f z = {0.f,0.f,0.f,0.f,0.f,0.f,0.f,0.f};
      acc[ni] = __builtin_amdgcn_wmma_f32_16x16x32_f16(false, Aq.h, false, Bk[ni].h,
                                                       (short)0, z, false, false);
    }

    // scale + relative bias + shift mask
#pragma unroll
    for (int ni = 0; ni < 4; ++ni)
#pragma unroll
      for (int r = 0; r < 8; ++r) {
        int i = mi * 16 + r + 8 * half;
        int j = ni * 16 + ln;
        int yi = i >> 3, xi = i & 7, yj = j >> 3, xj = j & 7;
        float v = acc[ni][r] * SC + pe[(yj - yi + 7) * 15 + (xj - xi + 7)];
        if (maskRow && ((yi >= DSP) != (yj >= DSP))) v = -1e30f;
        if (maskCol && ((xi >= DSP) != (xj >= DSP))) v = -1e30f;
        acc[ni][r] = v;
      }

    // row softmax (rows live in 16-lane halves; butterflies stay in-half)
#pragma unroll
    for (int r = 0; r < 8; ++r) {
      float m = fmaxf(fmaxf(acc[0][r], acc[1][r]), fmaxf(acc[2][r], acc[3][r]));
#pragma unroll
      for (int s = 1; s < 16; s <<= 1) m = fmaxf(m, __shfl_xor(m, s, 32));
      float sum = 0.f;
#pragma unroll
      for (int ni = 0; ni < 4; ++ni) {
        float e = __expf(acc[ni][r] - m);
        acc[ni][r] = e;
        sum += e;
      }
#pragma unroll
      for (int s = 1; s < 16; s <<= 1) sum += __shfl_xor(sum, s, 32);
      rinv[mi][r] = 1.0f / sum;
    }

    // exp(P) -> LDS, pitch 72 f16
#pragma unroll
    for (int ni = 0; ni < 4; ++ni)
#pragma unroll
      for (int r = 0; r < 8; ++r) {
        int i = mi * 16 + r + 8 * half;
        int j = ni * 16 + ln;
        P[i * 72 + j] = (_Float16)acc[ni][r];
      }
  }

  // --- P * V ---
  F16Frag Bv[2];
#pragma unroll
  for (int kc = 0; kc < 2; ++kc) {
    Bv[kc].u[0] = vz4(); Bv[kc].u[1] = vz4();
    if (ln < 8) {
      int kb_ = kc * 32 + half * 16;
#pragma unroll
      for (int i = 0; i < 16; ++i)
        Bv[kc].e[i] = Vs[(kb_ + i) * 32 + hc + ln];
    }
  }
#pragma unroll
  for (int mi = 0; mi < 4; ++mi) {
    v8f o = {0.f,0.f,0.f,0.f,0.f,0.f,0.f,0.f};
#pragma unroll
    for (int kc = 0; kc < 2; ++kc) {
      F16Frag Ap;
      int row = mi * 16 + ln;
      int kb_ = kc * 32 + half * 16;
      Ap.u[0] = *(const v4u*)(&P[row * 72 + kb_]);
      Ap.u[1] = *(const v4u*)(&P[row * 72 + kb_ + 8]);
      o = __builtin_amdgcn_wmma_f32_16x16x32_f16(false, Ap.h, false, Bv[kc].h,
                                                 (short)0, o, false, false);
    }
    // store (shifted frame), normalize by 1/rowsum
    if (ln < 8) {
#pragma unroll
      for (int r = 0; r < 8; ++r) {
        int i = mi * 16 + r + 8 * half;
        int wy = i >> 3, wx = i & 7;
        int y = wh * WSZ + wy, x = ww * WSZ + wx;
        float v = o[r] * rinv[mi][r];
        outb[((size_t)(b * IMH + y) * IMW + x) * 32 + hc + ln] = (_Float16)v;
      }
    }
  }
}

// ---------------------------------------------------------------------------
// Output conv (32 -> 3, SAME in the shifted frame) + roll back + residual.
// ---------------------------------------------------------------------------
__global__ void swin_outconv_kernel(const _Float16* __restrict__ att, // (B,H,W,32)
                                    const float* __restrict__ wout,   // (3,32,3,3)
                                    const float* __restrict__ bout,   // (3)
                                    const float* __restrict__ x0,     // (B,3,H,W)
                                    float* __restrict__ out)          // (B,3,H,W)
{
  __shared__ float ws[3 * 32 * 9];
  __shared__ float bs[3];
  for (int i = threadIdx.x; i < 3 * 32 * 9; i += blockDim.x) ws[i] = wout[i];
  if (threadIdx.x < 3) bs[threadIdx.x] = bout[threadIdx.x];
  __syncthreads();

  int idx = blockIdx.x * blockDim.x + threadIdx.x;
  if (idx >= BATCH * IMH * IMW) return;
  int xw = idx % IMW, yh = (idx / IMW) % IMH, b = idx / (IMW * IMH);
  int ys = (yh - DSP + IMH) % IMH;
  int xs = (xw - DSP + IMW) % IMW;

  float acc0 = bs[0], acc1 = bs[1], acc2 = bs[2];
  for (int ky = 0; ky < 3; ++ky) {
    int yy = ys + ky - 1;
    if (yy < 0 || yy >= IMH) continue;
    for (int kx = 0; kx < 3; ++kx) {
      int xx = xs + kx - 1;
      if (xx < 0 || xx >= IMW) continue;
      const _Float16* ap = att + ((size_t)(b * IMH + yy) * IMW + xx) * 32;
#pragma unroll
      for (int ci = 0; ci < 32; ++ci) {
        float v = (float)ap[ci];
        acc0 = fmaf(v, ws[((0 * 32 + ci) * 3 + ky) * 3 + kx], acc0);
        acc1 = fmaf(v, ws[((1 * 32 + ci) * 3 + ky) * 3 + kx], acc1);
        acc2 = fmaf(v, ws[((2 * 32 + ci) * 3 + ky) * 3 + kx], acc2);
      }
    }
  }
  size_t base = ((size_t)b * 3 * IMH + yh) * IMW + xw;
  out[base]                         = x0[base]                         + acc0;
  out[base + (size_t)IMH * IMW]     = x0[base + (size_t)IMH * IMW]     + acc1;
  out[base + (size_t)2 * IMH * IMW] = x0[base + (size_t)2 * IMH * IMW] + acc2;
}

// ---------------------------------------------------------------------------
extern "C" void kernel_launch(void* const* d_in, const int* in_sizes, int n_in,
                              void* d_out, int out_size, void* d_ws, size_t ws_size,
                              hipStream_t stream)
{
  (void)in_sizes; (void)n_in; (void)out_size; (void)ws_size;
  const float* x = (const float*)d_in[0];
  const float* w1[3]  = { (const float*)d_in[1],  (const float*)d_in[7],  (const float*)d_in[13] };
  const float* b1[3]  = { (const float*)d_in[2],  (const float*)d_in[8],  (const float*)d_in[14] };
  const float* w2[3]  = { (const float*)d_in[3],  (const float*)d_in[9],  (const float*)d_in[15] };
  const float* b2[3]  = { (const float*)d_in[4],  (const float*)d_in[10], (const float*)d_in[16] };
  const float* w3[3]  = { (const float*)d_in[5],  (const float*)d_in[11], (const float*)d_in[17] };
  const float* b3[3]  = { (const float*)d_in[6],  (const float*)d_in[12], (const float*)d_in[18] };
  const float* w_out  = (const float*)d_in[19];
  const float* b_out  = (const float*)d_in[20];
  const float* pos    = (const float*)d_in[21];
  float* out = (float*)d_out;

  // Workspace layout (bytes, all 1KB-aligned):
  const size_t SZ_H1    = (size_t)BATCH * IMH * IMW * 64 * 2;  // 37,748,736
  const size_t SZ_HWC32 = (size_t)BATCH * IMH * IMW * 32 * 2;  // 18,874,368
  char* ws = (char*)d_ws;
  _Float16* h1  = (_Float16*)(ws);
  _Float16* h2  = (_Float16*)(ws + SZ_H1);
  _Float16* qb  = (_Float16*)(ws + SZ_H1 + SZ_HWC32);
  _Float16* kb  = (_Float16*)(ws + SZ_H1 + 2 * SZ_HWC32);
  _Float16* vb  = (_Float16*)(ws + SZ_H1 + 3 * SZ_HWC32);
  _Float16* att = (_Float16*)(ws + SZ_H1 + 4 * SZ_HWC32);
  _Float16* wp2 = (_Float16*)(ws + SZ_H1 + 5 * SZ_HWC32);
  _Float16* wp3 = (_Float16*)(ws + SZ_H1 + 5 * SZ_HWC32 + 3 * 36 * 1024);
  _Float16* zpad = (_Float16*)(ws + SZ_H1 + 5 * SZ_HWC32 + 3 * 36 * 1024 + 3 * 18 * 1024);

  _Float16* qkv[3] = { qb, kb, vb };

  // Zero pad region (256 floats = 1KB) for branchless padding loads.
  swin_zero_kernel<<<1, 256, 0, stream>>>((float*)zpad);

  // Repack all weights (36 tiles for 64-ch conv2, 18 for 32-ch conv3).
  for (int br = 0; br < 3; ++br) {
    swin_repack_kernel<<<36, 32, 0, stream>>>(w2[br], 64, wp2 + (size_t)br * 36 * 512);
    swin_repack_kernel<<<18, 32, 0, stream>>>(w3[br], 32, wp3 + (size_t)br * 18 * 512);
  }

  // Branch conv stacks (h1/h2 reused: launches are stream-ordered).
  int pix_blocks = (BATCH * IMH * IMW + 255) / 256;
  dim3 cw_grid(IMW / 16 / 8, IMH, BATCH);   // 8 waves/block, one 16-px tile each
  for (int br = 0; br < 3; ++br) {
    swin_conv1_kernel<<<pix_blocks, 256, 0, stream>>>(x, w1[br], b1[br], h1);
    swin_convw_kernel<64, 1><<<cw_grid, 256, 0, stream>>>(
        h1, wp2 + (size_t)br * 36 * 512, b2[br], h2, zpad);
    swin_convw_kernel<32, 0><<<cw_grid, 256, 0, stream>>>(
        h2, wp3 + (size_t)br * 18 * 512, b3[br], qkv[br], zpad);
  }

  // Window attention: one block per window (4 waves = 4 heads).
  swin_attn_kernel<<<dim3(NWH * NWW, BATCH), 128, 0, stream>>>(qb, kb, vb, pos, att, zpad);

  // Output conv + roll-back + residual.
  swin_outconv_kernel<<<pix_blocks, 256, 0, stream>>>(att, w_out, b_out, x, out);
}
